// Qwen3Attention_59004260712880
// MI455X (gfx1250) — matrix operand order repaired and checked by reading, not compiled, not run on hardware
//
#include <hip/hip_runtime.h>

// ---------------- problem constants ----------------
#define B_   2
#define S_   2048
#define H_   2048
#define NH_  16
#define NKV_ 8
#define HD_  128
#define MROWS (B_ * S_)   // 4096

// ---------------- types ----------------
typedef __bf16 bf16_t;
typedef bf16_t        v16bf __attribute__((ext_vector_type(16)));
typedef float         v8f   __attribute__((ext_vector_type(8)));
typedef unsigned int  u32x4 __attribute__((ext_vector_type(4)));

union Frag32B { v16bf v; u32x4 q[2]; };

__device__ __forceinline__ unsigned short f2bf(float x) {
  unsigned int u = __float_as_uint(x);
  unsigned int r = (u + 0x7FFFu + ((u >> 16) & 1u)) >> 16;   // round-to-nearest-even
  return (unsigned short)r;
}

__device__ __forceinline__ v8f wmma_bf16(v16bf a, v16bf b, v8f c) {
  // D = A(16x32 bf16) * B(32x16 bf16) + C(16x16 f32)
  return __builtin_amdgcn_wmma_f32_16x16x32_bf16(false, a, false, b, (short)0, c, false, false);
}

// ---- CDNA5 async global->LDS copy (tracked by ASYNCcnt) ----
__device__ __forceinline__ void async_ld_b128(const void* g, void* lds) {
  unsigned int l = (unsigned int)(size_t)lds;                 // LDS_ADDR = flat[31:0]
  unsigned long long ga = (unsigned long long)(size_t)g;
  asm volatile("global_load_async_to_lds_b128 %0, %1, off"
               :: "v"(l), "v"(ga) : "memory");
}
template <int N>
__device__ __forceinline__ void wait_asynccnt() {
  asm volatile("s_wait_asynccnt %0" :: "n"(N) : "memory");
}

// ---------------- fp32 -> bf16 convert ----------------
__global__ __launch_bounds__(256) void cvt_kernel(const float* __restrict__ in,
                                                  unsigned short* __restrict__ out, int n) {
  int i = blockIdx.x * blockDim.x + threadIdx.x;
  if (i < n) out[i] = f2bf(in[i]);
}

// ---------------- V transpose: [B,S,NKV,HD] f32 -> [B,NKV,HD,S] bf16 ----------------
__global__ __launch_bounds__(256) void vtrans_kernel(const float* __restrict__ vf,
                                                     unsigned short* __restrict__ vt) {
  long idx = (long)blockIdx.x * blockDim.x + threadIdx.x;   // output-linear
  int s = (int)(idx % S_);
  long t = idx / S_;
  int d  = (int)(t % HD_);  t /= HD_;
  int hh = (int)(t % NKV_);
  int b  = (int)(t / NKV_);
  vt[idx] = f2bf(vf[(((long)b * S_ + s) * NKV_ + hh) * HD_ + d]);
}

// ---------------- RMSNorm + RoPE: [B,S,nh,HD] f32 -> [B,nh,S,HD] bf16 ----------------
// one wave per head-row; 4 elements per lane
__global__ __launch_bounds__(128) void normrope_kernel(const float* __restrict__ xin,
                                                       const float* __restrict__ wgt,
                                                       unsigned short* __restrict__ out,
                                                       int nheads) {
  const int wv   = threadIdx.x >> 5;
  const int lane = threadIdx.x & 31;
  const long rid = (long)blockIdx.x * 4 + wv;     // (b*S + s)*nheads + hh
  const int hh = (int)(rid % nheads);
  const long bs = rid / nheads;
  const int s = (int)(bs % S_);
  const int b = (int)(bs / S_);

  const float* x = xin + rid * HD_;
  float vv[4];
#pragma unroll
  for (int j = 0; j < 4; ++j) vv[j] = x[lane * 4 + j];
  float ss = vv[0]*vv[0] + vv[1]*vv[1] + vv[2]*vv[2] + vv[3]*vv[3];
#pragma unroll
  for (int off = 16; off >= 1; off >>= 1) ss += __shfl_xor(ss, off, 32);
  const float r = rsqrtf(ss * (1.0f / HD_) + 1e-6f);

#pragma unroll
  for (int j = 0; j < 4; ++j) {
    const int d = lane * 4 + j;
    const float xn = vv[j] * r * wgt[d];
    const int i = d & 63;                               // freq index (d mod 64)
    // inv_freq = theta^(-i/64) = exp(-i * ln(1e6)/64)
    const float ang = (float)s * __expf(-(float)i * (13.815510557964274f / 64.0f));
    const float cs = __cosf(ang);
    const float sn = __sinf(ang);
    const float partner = __shfl_xor(xn, 16, 32);       // element at d ^ 64
    const float rot = (d < 64) ? -partner : partner;    // rotate_half
    out[(((long)b * nheads + hh) * S_ + s) * HD_ + d] = f2bf(xn * cs + rot * sn);
  }
}

// ---------------- bf16 WMMA GEMM: C[M][N] = A[M][K] * Bm[N][K]^T ----------------
// 256 threads = 8 waves; tile 128x128, BK=32; wave -> 32x64 sub-tile (2x4 fragments)
// Double-buffered LDS filled with async global->LDS loads (ASYNCcnt pipelined).
__global__ __launch_bounds__(256) void gemm_bf16_kernel(const unsigned short* __restrict__ A,
                                                        const unsigned short* __restrict__ Bm,
                                                        float* __restrict__ C,
                                                        int M, int N, int K) {
  __shared__ unsigned short At[2][128 * 32];
  __shared__ unsigned short Bt[2][128 * 32];

  const int tid  = threadIdx.x;
  const int lane = tid & 31;
  const int w    = tid >> 5;
  const int wm   = w & 3;        // M sub-tile (32 rows)
  const int wn   = w >> 2;       // N sub-tile (64 cols)
  const int lg   = lane & 15;
  const int hs   = lane >> 4;    // lane-half select

  const int ntiles = N >> 7;
  const int mbase  = (blockIdx.x / ntiles) << 7;
  const int nbase  = (blockIdx.x % ntiles) << 7;

  v8f acc[2][4];
#pragma unroll
  for (int mi = 0; mi < 2; ++mi)
#pragma unroll
    for (int ni = 0; ni < 4; ++ni)
#pragma unroll
      for (int e = 0; e < 8; ++e) acc[mi][ni][e] = 0.0f;

  // staging assignment: 256 threads cover A(128x32) + B(128x32), 32B each side
  const int crow  = tid >> 1;    // 0..127
  const int chalf = tid & 1;     // which 16-element half of the 32-wide K chunk
  const unsigned short* gA = A  + (long)(mbase + crow) * K + chalf * 16;
  const unsigned short* gB = Bm + (long)(nbase + crow) * K + chalf * 16;
  unsigned short* lA0 = &At[0][crow * 32 + chalf * 16];
  unsigned short* lB0 = &Bt[0][crow * 32 + chalf * 16];

  const int niter = K >> 5;

  // prologue: stage k-chunk 0 into buffer 0
  async_ld_b128(gA,     lA0);
  async_ld_b128(gA + 8, lA0 + 8);
  async_ld_b128(gB,     lB0);
  async_ld_b128(gB + 8, lB0 + 8);

  for (int it = 0; it < niter; ++it) {
    const int cur = it & 1;
    if (it + 1 < niter) {    // stage next k-chunk into other buffer
      const int nk = (it + 1) << 5;
      unsigned short* lA = &At[cur ^ 1][crow * 32 + chalf * 16];
      unsigned short* lB = &Bt[cur ^ 1][crow * 32 + chalf * 16];
      async_ld_b128(gA + nk,     lA);
      async_ld_b128(gA + nk + 8, lA + 8);
      async_ld_b128(gB + nk,     lB);
      async_ld_b128(gB + nk + 8, lB + 8);
      wait_asynccnt<4>();    // current buffer's 4 loads complete (in-order)
    } else {
      wait_asynccnt<0>();
    }
    __syncthreads();

    Frag32B af[2];
#pragma unroll
    for (int mi = 0; mi < 2; ++mi) {
      const unsigned short* p = &At[cur][(wm * 32 + mi * 16 + lg) * 32];
      af[mi].q[0] = *(const u32x4*)(p + hs * 8);        // K = {0..7 | 8..15}
      af[mi].q[1] = *(const u32x4*)(p + 16 + hs * 8);   // K = {16..23 | 24..31}
    }
    Frag32B bfr[4];
#pragma unroll
    for (int ni = 0; ni < 4; ++ni) {
      const unsigned short* p = &Bt[cur][(wn * 64 + ni * 16 + lg) * 32];
      bfr[ni].q[0] = *(const u32x4*)(p + hs * 8);       // K = {0..7 | 8..15}
      bfr[ni].q[1] = *(const u32x4*)(p + 16 + hs * 8);  // K = {16..23 | 24..31}
    }
#pragma unroll
    for (int mi = 0; mi < 2; ++mi)
#pragma unroll
      for (int ni = 0; ni < 4; ++ni)
        acc[mi][ni] = wmma_bf16(af[mi].v, bfr[ni].v, acc[mi][ni]);
    __syncthreads();
  }

#pragma unroll
  for (int mi = 0; mi < 2; ++mi)
#pragma unroll
    for (int ni = 0; ni < 4; ++ni)
#pragma unroll
      for (int e = 0; e < 8; ++e) {
        const int row = mbase + wm * 32 + mi * 16 + (hs ? e + 8 : e);
        const int col = nbase + wn * 64 + ni * 16 + lg;
        C[(long)row * N + col] = acc[mi][ni][e];
      }
}

// ---------------- flash attention (causal, GQA) ----------------
// grid: (B*NH) * (S/64) blocks of 128 threads (4 waves); each wave: 16 q rows x 128 hd
// K / V^T tiles double-buffered in LDS via async global->LDS loads.
__global__ __launch_bounds__(128) void attn_kernel(const unsigned short* __restrict__ Qb,
                                                   const unsigned short* __restrict__ Kb,
                                                   const unsigned short* __restrict__ Vtb,
                                                   unsigned short* __restrict__ Ob) {
  __shared__ unsigned short Kt[2][32 * 128];   // [key][hd]
  __shared__ unsigned short Vt[2][128 * 32];   // [hd][key]
  __shared__ unsigned short Ps[4][16 * 32];    // per-wave P transpose scratch

  const int tid  = threadIdx.x;
  const int lane = tid & 31;
  const int w    = tid >> 5;
  const int lg   = lane & 15;
  const int hs   = lane >> 4;

  const int qtiles = S_ / 64;
  const int qb  = (blockIdx.x % qtiles) * 64;
  const int bh  = blockIdx.x / qtiles;
  const int b   = bh / NH_;
  const int h   = bh % NH_;
  const int kvh = h / (NH_ / NKV_);

  // persistent Q fragments (16 rows x 128 hd -> 4 A-frags)
  const int qrow = qb + w * 16 + lg;
  const unsigned short* qp = Qb + ((long)(b * NH_ + h) * S_ + qrow) * HD_;
  Frag32B qf[4];
#pragma unroll
  for (int c = 0; c < 4; ++c) {
    qf[c].q[0] = *(const u32x4*)(qp + c * 32 + hs * 8);
    qf[c].q[1] = *(const u32x4*)(qp + c * 32 + 16 + hs * 8);
  }

  v8f o[8];
  float m[8], l[8];
#pragma unroll
  for (int e = 0; e < 8; ++e) { m[e] = -1e30f; l[e] = 0.0f; }
#pragma unroll
  for (int hd = 0; hd < 8; ++hd)
#pragma unroll
    for (int e = 0; e < 8; ++e) o[hd][e] = 0.0f;

  const float sc = 0.08838834764831845f;  // 1/sqrt(128)
  const long kbbase = (long)(b * NKV_ + kvh) * S_;
  const long vtbase = (long)(b * NKV_ + kvh) * HD_;

  // staging assignment: K tile rows (tid/4, 32B part tid%4); V^T row tid (64B)
  const int krow  = tid >> 2;
  const int kpart = tid & 3;
  const unsigned short* gK = Kb  + (kbbase + krow) * HD_ + kpart * 32;
  const unsigned short* gV = Vtb + (vtbase + tid) * S_;

  const int niter = (qb + 64) >> 5;

  // prologue: stage key-chunk 0 into buffer 0 (6 async b128 per thread)
  {
    unsigned short* lK = &Kt[0][krow * 128 + kpart * 32];
    unsigned short* lV = &Vt[0][tid * 32];
    async_ld_b128(gK,     lK);
    async_ld_b128(gK + 8, lK + 8);
    async_ld_b128(gV,      lV);
    async_ld_b128(gV + 8,  lV + 8);
    async_ld_b128(gV + 16, lV + 16);
    async_ld_b128(gV + 24, lV + 24);
  }

  for (int it = 0; it < niter; ++it) {
    const int kt  = it << 5;
    const int cur = it & 1;
    if (it + 1 < niter) {   // stage next key chunk into other buffer
      const int nk = kt + 32;
      unsigned short* lK = &Kt[cur ^ 1][krow * 128 + kpart * 32];
      unsigned short* lV = &Vt[cur ^ 1][tid * 32];
      const unsigned short* gKn = gK + (long)nk * HD_;
      async_ld_b128(gKn,     lK);
      async_ld_b128(gKn + 8, lK + 8);
      const unsigned short* gVn = gV + nk;
      async_ld_b128(gVn,      lV);
      async_ld_b128(gVn + 8,  lV + 8);
      async_ld_b128(gVn + 16, lV + 16);
      async_ld_b128(gVn + 24, lV + 24);
      wait_asynccnt<6>();   // current buffer's 6 loads complete (in-order)
    } else {
      wait_asynccnt<0>();
    }
    __syncthreads();

    // scores: S = Q(16x128) @ K^T(128x32) -> two 16x16 fragments
    v8f s0, s1;
#pragma unroll
    for (int e = 0; e < 8; ++e) { s0[e] = 0.0f; s1[e] = 0.0f; }
#pragma unroll
    for (int c = 0; c < 4; ++c) {
      Frag32B b0, b1;
      const unsigned short* p0 = &Kt[cur][lg * 128 + c * 32];
      b0.q[0] = *(const u32x4*)(p0 + hs * 8); b0.q[1] = *(const u32x4*)(p0 + 16 + hs * 8);
      const unsigned short* p1 = &Kt[cur][(16 + lg) * 128 + c * 32];
      b1.q[0] = *(const u32x4*)(p1 + hs * 8); b1.q[1] = *(const u32x4*)(p1 + 16 + hs * 8);
      s0 = wmma_bf16(qf[c].v, b0.v, s0);
      s1 = wmma_bf16(qf[c].v, b1.v, s1);
    }

    // online softmax (rows = C-fragment slots, across 16-lane groups)
    float corr[8];
#pragma unroll
    for (int e = 0; e < 8; ++e) {
      const int row = qb + w * 16 + (hs ? e + 8 : e);
      float a0 = (kt + lg      <= row) ? s0[e] * sc : -1e30f;   // causal mask
      float a1 = (kt + 16 + lg <= row) ? s1[e] * sc : -1e30f;
      float mx = fmaxf(a0, a1);
#pragma unroll
      for (int off = 8; off >= 1; off >>= 1) mx = fmaxf(mx, __shfl_xor(mx, off, 16));
      const float mn = fmaxf(m[e], mx);
      const float cf = __expf(m[e] - mn);
      const float p0 = __expf(a0 - mn);
      const float p1 = __expf(a1 - mn);
      float rs = p0 + p1;
#pragma unroll
      for (int off = 8; off >= 1; off >>= 1) rs += __shfl_xor(rs, off, 16);
      l[e] = l[e] * cf + rs;
      m[e] = mn;
      corr[e] = cf;
      const int pr = hs ? e + 8 : e;
      Ps[w][pr * 32 + lg]      = f2bf(p0);      // transpose P via LDS
      Ps[w][pr * 32 + 16 + lg] = f2bf(p1);
    }
#pragma unroll
    for (int hd = 0; hd < 8; ++hd)
#pragma unroll
      for (int e = 0; e < 8; ++e) o[hd][e] *= corr[e];

    // P (16x32 bf16) as A-fragment
    Frag32B pf;
    const unsigned short* pp = &Ps[w][0] + lg * 32;
    pf.q[0] = *(const u32x4*)(pp + hs * 8);
    pf.q[1] = *(const u32x4*)(pp + 16 + hs * 8);

    // O += P @ V  (B-frags from V^T tile, K-contiguous)
#pragma unroll
    for (int hd = 0; hd < 8; ++hd) {
      Frag32B bv;
      const unsigned short* vp = &Vt[cur][(hd * 16 + lg) * 32];
      bv.q[0] = *(const u32x4*)(vp + hs * 8);
      bv.q[1] = *(const u32x4*)(vp + 16 + hs * 8);
      o[hd] = wmma_bf16(pf.v, bv.v, o[hd]);
    }
    __syncthreads();
  }

  // epilogue: divide by l, write bf16 attn [B,S,NH*HD]
#pragma unroll
  for (int hd = 0; hd < 8; ++hd)
#pragma unroll
    for (int e = 0; e < 8; ++e) {
      const int row = qb + w * 16 + (hs ? e + 8 : e);
      const float val = o[hd][e] / l[e];
      Ob[((long)b * S_ + row) * (NH_ * HD_) + h * HD_ + hd * 16 + lg] = f2bf(val);
    }
}

// ---------------- host launch ----------------
extern "C" void kernel_launch(void* const* d_in, const int* in_sizes, int n_in,
                              void* d_out, int out_size, void* d_ws, size_t ws_size,
                              hipStream_t stream) {
  (void)in_sizes; (void)n_in; (void)out_size; (void)ws_size;
  const float* X  = (const float*)d_in[0];
  // d_in[1] = attention_mask (causal, applied analytically in attn_kernel)
  const float* Wq = (const float*)d_in[2];
  const float* Wk = (const float*)d_in[3];
  const float* Wv = (const float*)d_in[4];
  const float* Wo = (const float*)d_in[5];
  const float* qw = (const float*)d_in[6];
  const float* kw = (const float*)d_in[7];

  char* ws = (char*)d_ws;
  size_t off = 0;
  auto alloc = [&](size_t bytes) -> char* {
    char* p = ws + off;
    off = (off + bytes + 255) & ~(size_t)255;
    return p;
  };

  unsigned short* Xb  = (unsigned short*)alloc((size_t)MROWS * H_ * 2);
  unsigned short* Wqb = (unsigned short*)alloc((size_t)(NH_ * HD_) * H_ * 2);
  unsigned short* Wkb = (unsigned short*)alloc((size_t)(NKV_ * HD_) * H_ * 2);
  unsigned short* Wvb = (unsigned short*)alloc((size_t)(NKV_ * HD_) * H_ * 2);
  unsigned short* Wob = (unsigned short*)alloc((size_t)H_ * (NH_ * HD_) * 2);
  float* qf = (float*)alloc((size_t)MROWS * NH_ * HD_ * 4);
  float* kf = (float*)alloc((size_t)MROWS * NKV_ * HD_ * 4);
  float* vf = (float*)alloc((size_t)MROWS * NKV_ * HD_ * 4);
  unsigned short* Qb  = (unsigned short*)alloc((size_t)B_ * NH_  * S_ * HD_ * 2);
  unsigned short* Kbn = (unsigned short*)alloc((size_t)B_ * NKV_ * S_ * HD_ * 2);
  unsigned short* Vtb = (unsigned short*)alloc((size_t)B_ * NKV_ * HD_ * S_ * 2);
  unsigned short* Ab  = (unsigned short*)alloc((size_t)MROWS * (NH_ * HD_) * 2);

  // 1) fp32 -> bf16 converts
  cvt_kernel<<<(MROWS * H_) / 256, 256, 0, stream>>>(X, Xb, MROWS * H_);
  cvt_kernel<<<(NH_ * HD_ * H_) / 256, 256, 0, stream>>>(Wq, Wqb, NH_ * HD_ * H_);
  cvt_kernel<<<(NKV_ * HD_ * H_) / 256, 256, 0, stream>>>(Wk, Wkb, NKV_ * HD_ * H_);
  cvt_kernel<<<(NKV_ * HD_ * H_) / 256, 256, 0, stream>>>(Wv, Wvb, NKV_ * HD_ * H_);
  cvt_kernel<<<(H_ * NH_ * HD_) / 256, 256, 0, stream>>>(Wo, Wob, H_ * NH_ * HD_);

  // 2) QKV projections (WMMA GEMMs)
  gemm_bf16_kernel<<<(MROWS / 128) * ((NH_ * HD_) / 128), 256, 0, stream>>>(Xb, Wqb, qf, MROWS, NH_ * HD_, H_);
  gemm_bf16_kernel<<<(MROWS / 128) * ((NKV_ * HD_) / 128), 256, 0, stream>>>(Xb, Wkb, kf, MROWS, NKV_ * HD_, H_);
  gemm_bf16_kernel<<<(MROWS / 128) * ((NKV_ * HD_) / 128), 256, 0, stream>>>(Xb, Wvb, vf, MROWS, NKV_ * HD_, H_);

  // 3) RMSNorm + RoPE (q, k) ; V transpose+convert
  normrope_kernel<<<(MROWS * NH_) / 4, 128, 0, stream>>>(qf, qw, Qb, NH_);
  normrope_kernel<<<(MROWS * NKV_) / 4, 128, 0, stream>>>(kf, kw, Kbn, NKV_);
  vtrans_kernel<<<((long)B_ * NKV_ * HD_ * S_) / 256, 256, 0, stream>>>(vf, Vtb);

  // 4) causal flash attention
  attn_kernel<<<B_ * NH_ * (S_ / 64), 128, 0, stream>>>(Qb, Kbn, Vtb, Ab);

  // 5) output projection -> d_out (fp32)
  gemm_bf16_kernel<<<(MROWS / 128) * (H_ / 128), 256, 0, stream>>>(Ab, Wob, (float*)d_out, MROWS, H_, NH_ * HD_);
}
